// LSTMDecoderWithAttention_20804821582369
// MI455X (gfx1250) — compile-verified, matrix operand order checked by reading, not run on hardware
//
#include <hip/hip_runtime.h>
#include <hip/hip_bf16.h>
#include <stdint.h>

#define B_  128
#define S_  512
#define E_  256   // EMBED == ENC == DEC == HID
#define H_  256
#define T_  256
#define G_  1024  // 4*HID
#define XK  512   // EMBED + ENC

typedef __attribute__((ext_vector_type(16))) __bf16 v16bf;
typedef __attribute__((ext_vector_type(8)))  float  v8f;

// ---------- scalar helpers ----------
__device__ __forceinline__ __bf16 f2bf(float f) {
  uint32_t u = __builtin_bit_cast(uint32_t, f);
  u += 0x7FFFu + ((u >> 16) & 1u);          // round-to-nearest-even
  uint16_t s = (uint16_t)(u >> 16);
  return __builtin_bit_cast(__bf16, s);
}
__device__ __forceinline__ float bf2f(__bf16 b) {
  uint16_t s = __builtin_bit_cast(uint16_t, b);
  uint32_t u = ((uint32_t)s) << 16;
  return __builtin_bit_cast(float, u);
}
__device__ __forceinline__ float fast_exp(float x) {
  return __builtin_amdgcn_exp2f(x * 1.4426950408889634f);  // v_exp_f32
}
__device__ __forceinline__ float fast_tanh(float x) {
  float e = fast_exp(2.0f * x);
  return 1.0f - 2.0f / (e + 1.0f);
}
__device__ __forceinline__ float fast_sigmoid(float x) {
  return 1.0f / (1.0f + fast_exp(-x));
}

// ---------- WMMA fragment helpers (CDNA5 16-bit A/B layout, ISA 7.12.2) ----------
// A (M x K) row-major: lane L<16 -> row L, K-groups {0..7},{16..23};
//                      lane L>=16 -> row L-16, K-groups {8..15},{24..31}.
// B operand for X @ W^T uses the same loader on W (N x K row-major).
__device__ __forceinline__ v16bf load_frag(const __bf16* __restrict__ p, int ld,
                                           int row0, int k0) {
  const int lane = threadIdx.x & 31;
  const int row  = row0 + (lane & 15);
  const int kg   = (lane >> 4) << 3;  // 0 or 8
  const __bf16* q = p + (size_t)row * ld + k0 + kg;
  v16bf a;
#pragma unroll
  for (int j = 0; j < 8; ++j) a[j] = q[j];
#pragma unroll
  for (int j = 0; j < 8; ++j) a[8 + j] = q[16 + j];
  return a;
}

template <int K>
__device__ __forceinline__ v8f wmma_xwt(const __bf16* __restrict__ X, int ldx,
                                        const __bf16* __restrict__ W, int ldw,
                                        int row0, int col0, v8f c) {
#pragma unroll
  for (int k = 0; k < K; k += 32) {
    v16bf a = load_frag(X, ldx, row0, k);
    v16bf b = load_frag(W, ldw, col0, k);
    c = __builtin_amdgcn_wmma_f32_16x16x32_bf16(false, a, false, b,
                                                (short)0, c, false, false);
  }
  return c;
}

// C/D 16x16 f32 layout: vgpr r, lane<16 -> (M=r, N=lane); lane>=16 -> (M=r+8, N=lane-16)
__device__ __forceinline__ void store_tile_f32(float* __restrict__ out, int ldo,
                                               int row0, int col0, v8f c,
                                               const float* __restrict__ bias) {
  const int lane = threadIdx.x & 31;
  const int n  = lane & 15;
  const int mb = (lane >> 4) << 3;
  const float bb = bias ? bias[col0 + n] : 0.0f;
#pragma unroll
  for (int r = 0; r < 8; ++r)
    out[(size_t)(row0 + mb + r) * ldo + col0 + n] = c[r] + bb;
}

// ---------- one-time prep kernels ----------
__global__ void k_cvt(const float* __restrict__ src, __bf16* __restrict__ dst, int n) {
  int i = blockIdx.x * blockDim.x + threadIdx.x;
  if (i < n) dst[i] = f2bf(src[i]);
}

__global__ void k_split_wattn(const float* __restrict__ Wa,
                              __bf16* __restrict__ Wh, __bf16* __restrict__ Wm) {
  int i = blockIdx.x * blockDim.x + threadIdx.x;  // 256*256
  int d = i >> 8, e = i & 255;
  Wh[i] = f2bf(Wa[d * 512 + e]);
  Wm[i] = f2bf(Wa[d * 512 + 256 + e]);
}

__global__ void k_bias(const float* __restrict__ bi, const float* __restrict__ bh,
                       float* __restrict__ bs) {
  int i = blockIdx.x * blockDim.x + threadIdx.x;
  if (i < G_) bs[i] = bi[i] + bh[i];
}

__global__ void k_init(float* __restrict__ c, __bf16* __restrict__ hbf,
                       float* __restrict__ out0) {
  int i = blockIdx.x * blockDim.x + threadIdx.x;  // B_*H_
  c[i] = 0.0f;
  hbf[i] = f2bf(0.0f);
  out0[i] = 0.0f;   // out[0] = zeros row
}

// memory (S,B,E) f32  ->  memT (B*S, E) bf16  (WMMA-friendly row-major A)
__global__ void k_memT(const float* __restrict__ mem, __bf16* __restrict__ memT) {
  int s = blockIdx.x, b = blockIdx.y, e = threadIdx.x;
  memT[((size_t)b * S_ + s) * E_ + e] = f2bf(mem[((size_t)s * B_ + b) * E_ + e]);
}

// mem_proj[(b*S+s), d] = memT @ Wm^T + b_attn   (bf16 output: keeps set << 192MB L2)
__global__ void k_memproj(const __bf16* __restrict__ memT, const __bf16* __restrict__ Wm,
                          const float* __restrict__ battn, __bf16* __restrict__ mp) {
  int wave = threadIdx.x >> 5;
  int row0 = blockIdx.x * 16;                 // grid.x = (B_*S_)/16 = 4096
  int col0 = blockIdx.y * 128 + wave * 16;    // grid.y = 2
  v8f c = {};
  c = wmma_xwt<E_>(memT, E_, Wm, E_, row0, col0, c);
  const int lane = threadIdx.x & 31;
  const int n = lane & 15, mb = (lane >> 4) << 3;
  const float bb = battn[col0 + n];
#pragma unroll
  for (int r = 0; r < 8; ++r)
    mp[(size_t)(row0 + mb + r) * E_ + col0 + n] = f2bf(c[r] + bb);
}

// ---------- per-step kernels ----------
__global__ void k_hwh(const __bf16* __restrict__ hbf, const __bf16* __restrict__ Wh,
                      float* __restrict__ hwh) {
  int wave = threadIdx.x >> 5;
  int row0 = blockIdx.x * 16;                 // grid (8, 2)
  int col0 = blockIdx.y * 128 + wave * 16;
  v8f c = {};
  c = wmma_xwt<E_>(hbf, E_, Wh, E_, row0, col0, c);
  store_tile_f32(hwh, E_, row0, col0, c, nullptr);
}

// scores[b,s] = sum_d tanh(mem_proj[b,s,d] + hwh[b,d])
__global__ void k_energy(const __bf16* __restrict__ mp, const float* __restrict__ hwh,
                         float* __restrict__ scores) {
  __shared__ float hs[E_];
  const int b = blockIdx.x, tid = threadIdx.x;
  hs[tid] = hwh[b * E_ + tid];
  __syncthreads();
  const int wave = tid >> 5, lane = tid & 31;
  const int s = blockIdx.y * 8 + wave;        // grid (B_, S_/8)
  const __bf16* row = mp + ((size_t)b * S_ + s) * E_ + lane * 8;
  float acc = 0.0f;
#pragma unroll
  for (int j = 0; j < 8; ++j)
    acc += fast_tanh(bf2f(row[j]) + hs[lane * 8 + j]);
#pragma unroll
  for (int off = 16; off > 0; off >>= 1) acc += __shfl_xor(acc, off, 32);
  if (lane == 0) scores[(size_t)b * S_ + s] = acc;
}

// softmax over S, ctx[b,e] = sum_s attn * memory[s,b,e]; builds bf16 x = [emb_t | ctx]
__global__ void k_attn_ctx(const float* __restrict__ scores, const float* __restrict__ mem,
                           const float* __restrict__ tgt_t, __bf16* __restrict__ xbf) {
  __shared__ float sm[S_];
  __shared__ float tmp8[8];
  const int b = blockIdx.x, tid = threadIdx.x;
  const int lane = tid & 31, wave = tid >> 5;

  float v0 = scores[(size_t)b * S_ + tid];
  float v1 = scores[(size_t)b * S_ + 256 + tid];

  float m = fmaxf(v0, v1);
#pragma unroll
  for (int off = 16; off > 0; off >>= 1) m = fmaxf(m, __shfl_xor(m, off, 32));
  if (lane == 0) tmp8[wave] = m;
  __syncthreads();
  m = fmaxf(fmaxf(fmaxf(tmp8[0], tmp8[1]), fmaxf(tmp8[2], tmp8[3])),
            fmaxf(fmaxf(tmp8[4], tmp8[5]), fmaxf(tmp8[6], tmp8[7])));

  float e0 = fast_exp(v0 - m), e1 = fast_exp(v1 - m);
  sm[tid] = e0;
  sm[tid + 256] = e1;
  float sum = e0 + e1;
#pragma unroll
  for (int off = 16; off > 0; off >>= 1) sum += __shfl_xor(sum, off, 32);
  __syncthreads();                       // protect tmp8 reuse + publish sm
  if (lane == 0) tmp8[wave] = sum;
  __syncthreads();
  const float inv = 1.0f / (tmp8[0] + tmp8[1] + tmp8[2] + tmp8[3] +
                            tmp8[4] + tmp8[5] + tmp8[6] + tmp8[7]);

  float acc = 0.0f;
#pragma unroll 4
  for (int s = 0; s < S_; ++s)
    acc += sm[s] * mem[((size_t)s * B_ + b) * E_ + tid];
  acc *= inv;

  xbf[(size_t)b * XK + E_ + tid] = f2bf(acc);                      // ctx half
  xbf[(size_t)b * XK + tid]      = f2bf(tgt_t[(size_t)b * E_ + tid]); // emb half
}

// gates = x @ Wih^T + h @ Whh^T + (b_ih + b_hh)
__global__ void k_gates(const __bf16* __restrict__ xbf, const __bf16* __restrict__ Wih,
                        const __bf16* __restrict__ hbf, const __bf16* __restrict__ Whh,
                        const float* __restrict__ bias, float* __restrict__ gates) {
  int wave = threadIdx.x >> 5;
  int row0 = blockIdx.x * 16;                 // grid (8, 8)
  int col0 = blockIdx.y * 128 + wave * 16;
  v8f c = {};
  c = wmma_xwt<XK>(xbf, XK, Wih, XK, row0, col0, c);
  c = wmma_xwt<H_>(hbf, H_, Whh, H_, row0, col0, c);
  store_tile_f32(gates, G_, row0, col0, c, bias);
}

__global__ void k_cell(const float* __restrict__ gates, float* __restrict__ cst,
                       __bf16* __restrict__ hbf, float* __restrict__ out_t) {
  const int b = blockIdx.x, d = threadIdx.x;
  const float* g = gates + (size_t)b * G_;
  float i  = fast_sigmoid(g[d]);
  float f  = fast_sigmoid(g[H_ + d]);
  float gg = fast_tanh(g[2 * H_ + d]);
  float o  = fast_sigmoid(g[3 * H_ + d]);
  float cn = f * cst[b * H_ + d] + i * gg;
  float hn = o * fast_tanh(cn);
  cst[b * H_ + d] = cn;
  hbf[b * H_ + d] = f2bf(hn);
  out_t[(size_t)b * H_ + d] = hn;
}

// ---------- host ----------
extern "C" void kernel_launch(void* const* d_in, const int* in_sizes, int n_in,
                              void* d_out, int out_size, void* d_ws, size_t ws_size,
                              hipStream_t stream) {
  const float* tgt   = (const float*)d_in[0];  // (T,B,E)
  const float* mem   = (const float*)d_in[1];  // (S,B,E)
  const float* Wattn = (const float*)d_in[2];  // (256,512)
  const float* battn = (const float*)d_in[3];  // (256)
  const float* Wih   = (const float*)d_in[4];  // (1024,512)
  const float* Whh   = (const float*)d_in[5];  // (1024,256)
  const float* bih   = (const float*)d_in[6];
  const float* bhh   = (const float*)d_in[7];
  float* out = (float*)d_out;                  // (T,B,H)

  char* p = (char*)d_ws;
  auto alloc = [&](size_t bytes) {
    char* r = p;
    p += (bytes + 255) & ~(size_t)255;
    return (void*)r;
  };
  __bf16* memT   = (__bf16*)alloc((size_t)B_ * S_ * E_ * 2);   // 33.5 MB
  __bf16* mproj  = (__bf16*)alloc((size_t)B_ * S_ * E_ * 2);   // 33.5 MB (L2-resident)
  __bf16* WhB    = (__bf16*)alloc((size_t)E_ * E_ * 2);
  __bf16* WmB    = (__bf16*)alloc((size_t)E_ * E_ * 2);
  __bf16* WihB   = (__bf16*)alloc((size_t)G_ * XK * 2);
  __bf16* WhhB   = (__bf16*)alloc((size_t)G_ * H_ * 2);
  float*  bias   = (float*)alloc((size_t)G_ * 4);
  __bf16* hbf    = (__bf16*)alloc((size_t)B_ * H_ * 2);
  float*  cst    = (float*)alloc((size_t)B_ * H_ * 4);
  float*  hwh    = (float*)alloc((size_t)B_ * E_ * 4);
  float*  scores = (float*)alloc((size_t)B_ * S_ * 4);
  __bf16* xbf    = (__bf16*)alloc((size_t)B_ * XK * 2);
  float*  gates  = (float*)alloc((size_t)B_ * G_ * 4);

  // one-time precompute
  k_split_wattn<<<(E_ * E_) / 256, 256, 0, stream>>>(Wattn, WhB, WmB);
  k_cvt<<<(G_ * XK + 255) / 256, 256, 0, stream>>>(Wih, WihB, G_ * XK);
  k_cvt<<<(G_ * H_ + 255) / 256, 256, 0, stream>>>(Whh, WhhB, G_ * H_);
  k_bias<<<4, 256, 0, stream>>>(bih, bhh, bias);
  k_init<<<(B_ * H_) / 256, 256, 0, stream>>>(cst, hbf, out);
  k_memT<<<dim3(S_, B_), 256, 0, stream>>>(mem, memT);
  k_memproj<<<dim3((B_ * S_) / 16, 2), 256, 0, stream>>>(memT, WmB, battn, mproj);

  // sequential scan over time
  for (int t = 1; t < T_; ++t) {
    k_hwh<<<dim3(8, 2), 256, 0, stream>>>(hbf, WhB, hwh);
    k_energy<<<dim3(B_, S_ / 8), 256, 0, stream>>>(mproj, hwh, scores);
    k_attn_ctx<<<B_, 256, 0, stream>>>(scores, mem, tgt + (size_t)t * B_ * E_, xbf);
    k_gates<<<dim3(8, 8), 256, 0, stream>>>(xbf, WihB, hbf, WhhB, bias, gates);
    k_cell<<<B_, 256, 0, stream>>>(gates, cst, hbf, out + (size_t)t * B_ * H_);
  }
}
